// GCN_1554778161831
// MI455X (gfx1250) — matrix operand order repaired and checked by reading, not compiled, not run on hardware
//
#include <hip/hip_runtime.h>

// ---------------------------------------------------------------------------
// Types for CDNA5 WMMA (wave32): v_wmma_f32_16x16x32_bf16
// ---------------------------------------------------------------------------
typedef __bf16 v16bf __attribute__((ext_vector_type(16)));
typedef float  v8f   __attribute__((ext_vector_type(8)));

struct Frag32B { uint4 lo, hi; };   // 32 bytes, bit-cast to v16bf

#define N_NODES   20000
#define N_EDGES   160000
#define N_GRAPHS  64
#define IN_DIM    128
#define HID       512
#define N_CLASSES 16

#define BM 128
#define BN 64
#define BK 32
#define GEMM_THREADS 256

__device__ __forceinline__ unsigned short f32_to_bf16(float f) {
    union { float f; unsigned int u; } v; v.f = f;
    unsigned int u = v.u;
    unsigned int r = u + 0x7FFFu + ((u >> 16) & 1u);  // round-to-nearest-even
    return (unsigned short)(r >> 16);
}

// ---------------------------------------------------------------------------
// Tiled WMMA GEMM core. A is [M][K] bf16 row-major; Bt is the weight
// PRE-TRANSPOSED to [N][K] bf16 so both LDS tiles stage with b128 ops.
// 256 threads = 8 waves; block tile 128x64, K-step 32; each wave owns a
// 16-row slice and 4 16x16 f32 accumulators. K must be a multiple of 32.
// INTERIOR=true: no bounds guards anywhere (fast path).
// ---------------------------------------------------------------------------
template <bool INTERIOR>
__device__ __forceinline__ void gemm_core(
    const unsigned short* __restrict__ A,
    const unsigned short* __restrict__ Bt,
    float* __restrict__ C,
    const float* __restrict__ bias,
    int M, int N, int K, int doRelu,
    unsigned short* sA, unsigned short* sB,
    int rowBase, int colBase)
{
    constexpr int LDA = BK + 8;                 // 40 ushorts = 80 B rows (16B aligned)
    const int tid    = threadIdx.x;
    const int wave   = tid >> 5;
    const int lane   = tid & 31;
    const int laneLo = lane & 15;
    const bool hiHalf = lane >= 16;

    v8f acc[BN / 16] = {};

    // staging coordinates (fixed per thread)
    const int ar0 = (tid)                 >> 2;   // A rows 0..63
    const int ar1 = (tid + GEMM_THREADS)  >> 2;   // A rows 64..127
    const int akg = (tid & 3) * 8;                // k-group 0/8/16/24
    const int bn  = tid >> 2;                     // B row (= output col) 0..63
    // fragment coordinates
    const int far  = wave * 16 + laneLo;          // A fragment row in tile
    const int fakb = hiHalf ? 8 : 0;              // ISA 7.12.2 A layout k-base
    const int fbkb = hiHalf ? 8 : 0;              // B (col-per-lane) k-base, same as A

    for (int k0 = 0; k0 < K; k0 += BK) {
        // ---- stage A: 128x32 bf16 tile ----
        {
            uint4 v0, v1;
            if (INTERIOR) {
                v0 = *(const uint4*)(A + (size_t)(rowBase + ar0) * K + k0 + akg);
                v1 = *(const uint4*)(A + (size_t)(rowBase + ar1) * K + k0 + akg);
                if (k0 + BK < K) {   // prefetch next A tile -> global_prefetch_b8
                    __builtin_prefetch(A + (size_t)(rowBase + ar0) * K + k0 + BK + akg, 0, 1);
                    __builtin_prefetch(A + (size_t)(rowBase + ar1) * K + k0 + BK + akg, 0, 1);
                }
            } else {
                v0 = make_uint4(0u, 0u, 0u, 0u);
                v1 = make_uint4(0u, 0u, 0u, 0u);
                if (rowBase + ar0 < M)
                    v0 = *(const uint4*)(A + (size_t)(rowBase + ar0) * K + k0 + akg);
                if (rowBase + ar1 < M)
                    v1 = *(const uint4*)(A + (size_t)(rowBase + ar1) * K + k0 + akg);
            }
            *(uint4*)(&sA[ar0 * LDA + akg]) = v0;
            *(uint4*)(&sA[ar1 * LDA + akg]) = v1;
        }
        // ---- stage B: 64(cols) x 32(k) from pre-transposed Bt[N][K] ----
        {
            uint4 v0;
            if (INTERIOR) {
                v0 = *(const uint4*)(Bt + (size_t)(colBase + bn) * K + k0 + akg);
            } else {
                v0 = make_uint4(0u, 0u, 0u, 0u);
                if (colBase + bn < N)
                    v0 = *(const uint4*)(Bt + (size_t)(colBase + bn) * K + k0 + akg);
            }
            *(uint4*)(&sB[bn * LDA + akg]) = v0;
        }
        __syncthreads();

        // ---- A fragment: lanes 0-15 K={0..7,16..23}; lanes 16-31 K={8..15,24..31}
        Frag32B fa;
        fa.lo = *(const uint4*)(&sA[far * LDA + fakb]);
        fa.hi = *(const uint4*)(&sA[far * LDA + fakb + 16]);
        v16bf afrag = __builtin_bit_cast(v16bf, fa);

        #pragma unroll
        for (int j = 0; j < BN / 16; ++j) {
            // B fragment: col-per-lane; lanes 0-15 K={0..7,16..23}; lanes 16-31 K={8..15,24..31}
            const int fbn = j * 16 + laneLo;
            Frag32B fb;
            fb.lo = *(const uint4*)(&sB[fbn * LDA + fbkb]);
            fb.hi = *(const uint4*)(&sB[fbn * LDA + fbkb + 16]);
            v16bf bfrag = __builtin_bit_cast(v16bf, fb);
            acc[j] = __builtin_amdgcn_wmma_f32_16x16x32_bf16(
                false, afrag, false, bfrag, (short)0, acc[j], false, false);
        }
        __syncthreads();
    }

    // ---- epilogue: D layout — VGPR v, lanes 0-15: M=v; lanes 16-31: M=v+8 ----
    #pragma unroll
    for (int j = 0; j < BN / 16; ++j) {
        int col = colBase + j * 16 + laneLo;
        if (INTERIOR || col < N) {
            float bv = bias ? bias[col] : 0.0f;
            #pragma unroll
            for (int v = 0; v < 8; ++v) {
                int row = rowBase + wave * 16 + v + (hiHalf ? 8 : 0);
                if (INTERIOR || row < M) {
                    float val = acc[j][v] + bv;
                    if (doRelu) val = fmaxf(val, 0.0f);
                    C[(size_t)row * N + col] = val;
                }
            }
        }
    }
}

__global__ __launch_bounds__(GEMM_THREADS)
void wmma_gemm_bf16(const unsigned short* __restrict__ A,
                    const unsigned short* __restrict__ Bt,   // [N][K] transposed
                    float* __restrict__ C,
                    const float* __restrict__ bias,
                    int M, int N, int K, int doRelu)
{
    constexpr int LDA = BK + 8;
    __shared__ __align__(16) unsigned short sA[BM * LDA];
    __shared__ __align__(16) unsigned short sB[BN * LDA];

    const int rowBase = blockIdx.y * BM;
    const int colBase = blockIdx.x * BN;

    // Block-uniform branch: interior blocks take the guard-free fast path.
    if (rowBase + BM <= M && colBase + BN <= N)
        gemm_core<true >(A, Bt, C, bias, M, N, K, doRelu, sA, sB, rowBase, colBase);
    else
        gemm_core<false>(A, Bt, C, bias, M, N, K, doRelu, sA, sB, rowBase, colBase);
}

// ---------------------------------------------------------------------------
// Elementwise / graph kernels
// ---------------------------------------------------------------------------
__global__ void zero_f32(float* __restrict__ p, long n) {
    long i = (long)blockIdx.x * blockDim.x + threadIdx.x;
    long stride = (long)gridDim.x * blockDim.x;
    for (; i < n; i += stride) p[i] = 0.0f;
}

__global__ void degree_kernel(const int* __restrict__ src, const int* __restrict__ dst,
                              float* __restrict__ dout, float* __restrict__ din, int E) {
    int e = blockIdx.x * blockDim.x + threadIdx.x;
    if (e < E) {
        atomicAdd(&dout[src[e]], 1.0f);
        atomicAdd(&din[dst[e]], 1.0f);
    }
}

__global__ void rsqrt_kernel(const float* __restrict__ deg, float* __restrict__ out, int n) {
    int i = blockIdx.x * blockDim.x + threadIdx.x;
    if (i < n) out[i] = rsqrtf(fmaxf(deg[i], 1.0f));
}

// hbf[n,c] = bf16( h[n,c] * scale[n] )
__global__ void scale_to_bf16(const float* __restrict__ h, const float* __restrict__ s,
                              unsigned short* __restrict__ out, int C, long total) {
    long i = (long)blockIdx.x * blockDim.x + threadIdx.x;
    if (i < total) {
        long node = i / C;
        out[i] = f32_to_bf16(h[i] * s[node]);
    }
}

__global__ void f32_to_bf16_kernel(const float* __restrict__ in,
                                   unsigned short* __restrict__ out, long n) {
    long i = (long)blockIdx.x * blockDim.x + threadIdx.x;
    if (i < n) out[i] = f32_to_bf16(in[i]);
}

// Wt[n*K + k] = bf16( W[k*N + n] )  — convert + transpose weight to [N][K]
__global__ void wT_to_bf16_kernel(const float* __restrict__ W,
                                  unsigned short* __restrict__ Wt, int K, int N) {
    long i = (long)blockIdx.x * blockDim.x + threadIdx.x;
    if (i < (long)K * N) {
        int n = (int)(i / K);
        int k = (int)(i % K);
        Wt[i] = f32_to_bf16(W[(long)k * N + n]);
    }
}

// agg[dst[e], :] += proj[src[e], :]  (one block per edge, 128 thr x 4 floats)
__global__ void scatter_kernel(const float* __restrict__ proj,
                               const int* __restrict__ src, const int* __restrict__ dst,
                               float* __restrict__ agg) {
    int e = blockIdx.x;
    int s = src[e], d = dst[e];
    int c = threadIdx.x * 4;
    float4 v = *(const float4*)(proj + (size_t)s * HID + c);
    float* a = agg + (size_t)d * HID + c;
    atomicAdd(a + 0, v.x);
    atomicAdd(a + 1, v.y);
    atomicAdd(a + 2, v.z);
    atomicAdd(a + 3, v.w);
}

// h[n,c] = relu(agg[n,c] * din[n] + b[c])
__global__ void finalize_kernel(const float* __restrict__ agg, const float* __restrict__ din,
                                const float* __restrict__ b, float* __restrict__ h,
                                long total) {
    long i = (long)blockIdx.x * blockDim.x + threadIdx.x;
    if (i < total) {
        long node = i / HID;
        int  col  = (int)(i % HID);
        h[i] = fmaxf(agg[i] * din[node] + b[col], 0.0f);
    }
}

// per-graph sum + count (one block per node)
__global__ void pool_acc(const float* __restrict__ h, const int* __restrict__ gid,
                         float* __restrict__ sums, float* __restrict__ cnt) {
    int n = blockIdx.x;
    int g = gid[n];
    int c = threadIdx.x * 4;
    float4 v = *(const float4*)(h + (size_t)n * HID + c);
    float* s = sums + (size_t)g * HID + c;
    atomicAdd(s + 0, v.x);
    atomicAdd(s + 1, v.y);
    atomicAdd(s + 2, v.z);
    atomicAdd(s + 3, v.w);
    if (threadIdx.x == 0) atomicAdd(&cnt[g], 1.0f);
}

__global__ void pool_div(const float* __restrict__ sums, const float* __restrict__ cnt,
                         float* __restrict__ hg, long total) {
    long i = (long)blockIdx.x * blockDim.x + threadIdx.x;
    if (i < total) {
        long g = i / HID;
        hg[i] = sums[i] / fmaxf(cnt[g], 1.0f);
    }
}

// ---------------------------------------------------------------------------
// Host launch
// ---------------------------------------------------------------------------
static inline dim3 gs(long n, int t) { return dim3((unsigned)((n + t - 1) / t)); }

extern "C" void kernel_launch(void* const* d_in, const int* in_sizes, int n_in,
                              void* d_out, int out_size, void* d_ws, size_t ws_size,
                              hipStream_t stream) {
    const float* x    = (const float*)d_in[0];
    const int*   src  = (const int*)d_in[1];
    const int*   dst  = (const int*)d_in[2];
    const int*   gid  = (const int*)d_in[3];
    const float* W1   = (const float*)d_in[4];  const float* b1  = (const float*)d_in[5];
    const float* W2   = (const float*)d_in[6];  const float* b2  = (const float*)d_in[7];
    const float* W3   = (const float*)d_in[8];  const float* b3  = (const float*)d_in[9];
    const float* Wc1  = (const float*)d_in[10]; const float* bc1 = (const float*)d_in[11];
    const float* Wc2  = (const float*)d_in[12]; const float* bc2 = (const float*)d_in[13];
    const float* Wc3  = (const float*)d_in[14]; const float* bc3 = (const float*)d_in[15];

    // ---- carve workspace ----
    char* p = (char*)d_ws;
    auto carve = [&](size_t bytes) -> void* {
        void* r = (void*)p;
        p += (bytes + 255) & ~(size_t)255;
        return r;
    };
    float*          deg_out = (float*)carve(N_NODES * 4);
    float*          deg_in  = (float*)carve(N_NODES * 4);
    float*          dis     = (float*)carve(N_NODES * 4);   // dout_isqrt
    float*          nis     = (float*)carve(N_NODES * 4);   // din_isqrt
    float*          h       = (float*)carve((size_t)N_NODES * HID * 4);
    unsigned short* hbf     = (unsigned short*)carve((size_t)N_NODES * HID * 2);
    float*          proj    = (float*)carve((size_t)N_NODES * HID * 4);
    float*          agg     = (float*)carve((size_t)N_NODES * HID * 4);
    unsigned short* Wbf     = (unsigned short*)carve((size_t)HID * HID * 2);  // [N][K]
    float*          sums    = (float*)carve((size_t)N_GRAPHS * HID * 4);
    float*          cnt     = (float*)carve(N_GRAPHS * 4);
    float*          hg      = (float*)carve((size_t)N_GRAPHS * HID * 4);
    unsigned short* hgbf    = (unsigned short*)carve((size_t)N_GRAPHS * HID * 2);
    float*          g1      = (float*)carve((size_t)N_GRAPHS * HID * 4);
    unsigned short* g1bf    = (unsigned short*)carve((size_t)N_GRAPHS * HID * 2);
    float*          g2      = (float*)carve((size_t)N_GRAPHS * HID * 4);
    unsigned short* g2bf    = (unsigned short*)carve((size_t)N_GRAPHS * HID * 2);

    const long NH = (long)N_NODES * HID;

    // ---- degrees + rsqrt norms ----
    zero_f32<<<gs(N_NODES, 256), 256, 0, stream>>>(deg_out, N_NODES);
    zero_f32<<<gs(N_NODES, 256), 256, 0, stream>>>(deg_in,  N_NODES);
    degree_kernel<<<gs(N_EDGES, 256), 256, 0, stream>>>(src, dst, deg_out, deg_in, N_EDGES);
    rsqrt_kernel<<<gs(N_NODES, 256), 256, 0, stream>>>(deg_out, dis, N_NODES);
    rsqrt_kernel<<<gs(N_NODES, 256), 256, 0, stream>>>(deg_in,  nis, N_NODES);

    // ---- one graph-conv layer ----
    auto conv_layer = [&](const float* h_in, int Din, const float* W, const float* b) {
        long nin = (long)N_NODES * Din;
        scale_to_bf16<<<gs(nin, 256), 256, 0, stream>>>(h_in, dis, hbf, Din, nin);
        wT_to_bf16_kernel<<<gs((long)Din * HID, 256), 256, 0, stream>>>(W, Wbf, Din, HID);
        {
            dim3 grid((HID + BN - 1) / BN, (N_NODES + BM - 1) / BM);
            wmma_gemm_bf16<<<grid, GEMM_THREADS, 0, stream>>>(
                hbf, Wbf, proj, nullptr, N_NODES, HID, Din, 0);
        }
        zero_f32<<<4096, 256, 0, stream>>>(agg, NH);
        scatter_kernel<<<N_EDGES, HID / 4, 0, stream>>>(proj, src, dst, agg);
        finalize_kernel<<<gs(NH, 256), 256, 0, stream>>>(agg, nis, b, h, NH);
    };

    conv_layer(x, IN_DIM, W1, b1);
    conv_layer(h, HID,   W2, b2);
    conv_layer(h, HID,   W3, b3);

    // ---- mean pooling over graphs ----
    const long GH = (long)N_GRAPHS * HID;
    zero_f32<<<gs(GH, 256), 256, 0, stream>>>(sums, GH);
    zero_f32<<<1, 64, 0, stream>>>(cnt, N_GRAPHS);
    pool_acc<<<N_NODES, HID / 4, 0, stream>>>(h, gid, sums, cnt);
    pool_div<<<gs(GH, 256), 256, 0, stream>>>(sums, cnt, hg, GH);

    // ---- MLP head ----
    f32_to_bf16_kernel<<<gs(GH, 256), 256, 0, stream>>>(hg, hgbf, GH);

    wT_to_bf16_kernel<<<gs((long)HID * HID, 256), 256, 0, stream>>>(Wc1, Wbf, HID, HID);
    {
        dim3 grid((HID + BN - 1) / BN, (N_GRAPHS + BM - 1) / BM);
        wmma_gemm_bf16<<<grid, GEMM_THREADS, 0, stream>>>(
            hgbf, Wbf, g1, bc1, N_GRAPHS, HID, HID, 1);
    }
    f32_to_bf16_kernel<<<gs(GH, 256), 256, 0, stream>>>(g1, g1bf, GH);

    wT_to_bf16_kernel<<<gs((long)HID * HID, 256), 256, 0, stream>>>(Wc2, Wbf, HID, HID);
    {
        dim3 grid((HID + BN - 1) / BN, (N_GRAPHS + BM - 1) / BM);
        wmma_gemm_bf16<<<grid, GEMM_THREADS, 0, stream>>>(
            g1bf, Wbf, g2, bc2, N_GRAPHS, HID, HID, 1);
    }
    f32_to_bf16_kernel<<<gs(GH, 256), 256, 0, stream>>>(g2, g2bf, GH);

    wT_to_bf16_kernel<<<gs((long)HID * N_CLASSES, 256), 256, 0, stream>>>(Wc3, Wbf, HID, N_CLASSES);
    {
        dim3 grid((N_CLASSES + BN - 1) / BN, (N_GRAPHS + BM - 1) / BM);
        wmma_gemm_bf16<<<grid, GEMM_THREADS, 0, stream>>>(
            g2bf, Wbf, (float*)d_out, bc3, N_GRAPHS, N_CLASSES, HID, 0);
    }
}
